// VectorQuantizer_10986526343950
// MI455X (gfx1250) — matrix operand order repaired and checked by reading, not compiled, not run on hardware
//
#include <hip/hip_runtime.h>
#include <hip/hip_bf16.h>

typedef __attribute__((ext_vector_type(16))) __bf16 v16bf;
typedef __attribute__((ext_vector_type(8)))  float  v8f;

#define NE    1024      // codebook entries
#define EDIM  256       // embedding dim
#define HW    4096      // 64*64 tokens per batch image
#define NB    16        // batch
#define NTOK  (NB*HW)   // 65536 tokens
#define ZQ_ELEMS (NB*EDIM*HW) // 16777216

// round-to-nearest-even fp32 -> bf16
__device__ __forceinline__ unsigned short f2bf(float f) {
    unsigned int u = __float_as_uint(f);
    u += 0x7FFFu + ((u >> 16) & 1u);
    return (unsigned short)(u >> 16);
}

// ---------------------------------------------------------------------------
// Prep: ||e_n||^2 and pack emb into bf16 B-fragments.
// B tile (32x16, K-major) lane layout used here (contiguous k, same as A): lanes 0-15 hold K=0..15
// (j=K) for N=lane; lanes 16-31 hold K=16..31 (j=K-16) for N=lane-16.
// Storage: Bp[((nt*8 + ks)*32 + lane)*16 + j], nt=n/16, ks=k/32.
// ---------------------------------------------------------------------------
__global__ void vq_prep(const float* __restrict__ emb,
                        unsigned short* __restrict__ Bp,
                        float* __restrict__ en2) {
    int n = blockIdx.x * blockDim.x + threadIdx.x;   // 0..1023
    int nt = n >> 4, r = n & 15;
    float s = 0.0f;
    for (int k = 0; k < EDIM; ++k) {
        float v = emb[n * EDIM + k];
        s += v * v;
        int ks   = k >> 5;
        int kk   = k & 31;
        int lane = r + ((kk >> 4) << 4);
        int j    = kk & 15;
        Bp[(((nt * 8 + ks) * 32 + lane) << 4) + j] = f2bf(v);
    }
    en2[n] = s;
}

// ---------------------------------------------------------------------------
// Main: per block (4 waves), 128 tokens x 1024 codes.
// Each wave owns 2 sub-tiles of 16 rows -> every B fragment feeds 2 WMMAs.
// ---------------------------------------------------------------------------
__global__ __launch_bounds__(128) void vq_main(const float* __restrict__ z,
                                               const unsigned short* __restrict__ Bp,
                                               const float* __restrict__ en2,
                                               int* __restrict__ idx_i,
                                               float* __restrict__ idx_f) {
    // A fragments: [ks(8)][sub(8)][lane(32)][j(16)] bf16 = 64 KB
    __shared__ __align__(32) unsigned short Afrag[8 * 8 * 32 * 16];

    const int tid = threadIdx.x;
    const int wg  = blockIdx.x;          // 0..511
    const int b   = wg >> 5;             // batch image (32 tiles per image)
    const int t0  = (wg & 31) << 7;      // token base within image (tiles of 128)
    const float* zb = z + (size_t)b * (EDIM * HW);

    // Stage A tile (128 tokens x 256 channels) into fragment-swizzled LDS.
    // z is NCHW: (token m, channel k) at zb[k*HW + t0 + m] -> coalesced in m.
    // A 16x32 bf16 layout: lane r: j=0..15 <-> K=0..15;
    // lane r+16: j=0..15 <-> K=16..31.
    for (int i = 0; i < 256; ++i) {
        int e = i * 128 + tid;           // 32768 elements total
        int k = e >> 7;                  // channel 0..255
        int m = e & 127;                 // token row 0..127
        float v = zb[(size_t)k * HW + t0 + m];
        int ks = k >> 5, kk = k & 31;
        int s  = m >> 4, r  = m & 15;
        int grp  = kk >> 3;
        int lane = r + ((kk >> 4) << 4);
        int j    = kk & 15;
        Afrag[(((ks * 8 + s) * 32 + lane) << 4) + j] = f2bf(v);
    }
    __syncthreads();

    const int wave = tid >> 5;
    const int lane = tid & 31;
    const int s0 = wave << 1;            // this wave's two sub-tiles: s0, s0+1

    float bestv0[8], bestv1[8];
    int   besti0[8], besti1[8];
#pragma unroll
    for (int i = 0; i < 8; ++i) {
        bestv0[i] = 3.4e38f; besti0[i] = 0;
        bestv1[i] = 3.4e38f; besti1[i] = 0;
    }

    for (int nt = 0; nt < 64; ++nt) {          // 16-code tiles
        // pull next nt's 8KB B block into the WGP-level cache one iteration
        // early (locality 3 -> WGP scope: fill all cache levels on miss)
        {
            int ntn = (nt + 1) & 63;
            __builtin_prefetch(&Bp[(ntn * 8 * 32) << 4] + lane * 128, 0, 3);
        }
        v8f acc0 = {}, acc1 = {};
#pragma unroll
        for (int ks = 0; ks < 8; ++ks) {       // full K = 256
            v16bf bm = *(const v16bf*)&Bp[(((nt * 8 + ks) * 32 + lane) << 4)];
            v16bf a0 = *(const v16bf*)&Afrag[(((ks * 8 + s0) * 32 + lane) << 4)];
            v16bf a1 = *(const v16bf*)&Afrag[(((ks * 8 + s0 + 1) * 32 + lane) << 4)];
            acc0 = __builtin_amdgcn_wmma_f32_16x16x32_bf16(
                false, a0, false, bm, (short)0, acc0, false, false);
            acc1 = __builtin_amdgcn_wmma_f32_16x16x32_bf16(
                false, a1, false, bm, (short)0, acc1, false, false);
        }
        // C layout: slot i -> row i (lanes 0-15) / row i+8 (lanes 16-31), N = lane%16
        int   n  = (nt << 4) + (lane & 15);
        float e2 = en2[n];
#pragma unroll
        for (int i = 0; i < 8; ++i) {
            float d0 = e2 - 2.0f * acc0[i];    // ||z||^2 dropped (constant per row)
            float d1 = e2 - 2.0f * acc1[i];
            bool b0 = d0 < bestv0[i];          // strict: keep earliest n
            bool b1 = d1 < bestv1[i];
            besti0[i] = b0 ? n : besti0[i];  bestv0[i] = b0 ? d0 : bestv0[i];
            besti1[i] = b1 ? n : besti1[i];  bestv1[i] = b1 ? d1 : bestv1[i];
        }
    }

    // Min-reduce across the 16 lanes sharing each row (tie -> lower index).
#pragma unroll
    for (int p = 0; p < 2; ++p) {
#pragma unroll
        for (int i = 0; i < 8; ++i) {
            float v  = p ? bestv1[i] : bestv0[i];
            int   ix = p ? besti1[i] : besti0[i];
            for (int off = 8; off; off >>= 1) {
                float ov = __shfl_xor(v, off, 16);
                int   oi = __shfl_xor(ix, off, 16);
                if (ov < v || (ov == v && oi < ix)) { v = ov; ix = oi; }
            }
            if ((lane & 15) == 0) {
                int row = i + ((lane >> 4) << 3);        // lane0 -> i, lane16 -> i+8
                int tok = b * HW + t0 + ((s0 + p) << 4) + row;
                idx_i[tok] = ix;
                idx_f[tok] = (float)ix;
            }
        }
    }
}

// ---------------------------------------------------------------------------
// Gather: z_q[b,c,h,w] = emb[idx[b,h,w], c], coalesced NCHW writes.
// ---------------------------------------------------------------------------
__global__ void vq_gather(const float* __restrict__ emb,
                          const int* __restrict__ idx_i,
                          float* __restrict__ zq) {
    unsigned int e = blockIdx.x * blockDim.x + threadIdx.x;   // < 16777216
    int t = e & (HW - 1);
    int c = (e >> 12) & (EDIM - 1);
    int b = e >> 20;
    int tok = (b << 12) + t;
    zq[e] = emb[idx_i[tok] * EDIM + c];
}

extern "C" void kernel_launch(void* const* d_in, const int* in_sizes, int n_in,
                              void* d_out, int out_size, void* d_ws, size_t ws_size,
                              hipStream_t stream) {
    const float* z   = (const float*)d_in[0];   // (16,256,64,64) fp32
    const float* emb = (const float*)d_in[1];   // (1024,256) fp32

    // workspace: [Bp bf16 512KB][en2 4KB][idx int 256KB]
    unsigned short* Bp  = (unsigned short*)d_ws;
    float*          en2 = (float*)((char*)d_ws + 512 * 1024);
    int*            idx = (int*)((char*)d_ws + 512 * 1024 + 4096);

    float* zq    = (float*)d_out;               // 16777216 floats
    float* idx_f = (float*)d_out + ZQ_ELEMS;    // 65536 indices (exact in fp32)

    vq_prep<<<dim3(NE / 256), dim3(256), 0, stream>>>(emb, Bp, en2);
    vq_main<<<dim3(NTOK / 128), dim3(128), 0, stream>>>(z, Bp, en2, idx, idx_f);
    vq_gather<<<dim3(ZQ_ELEMS / 256), dim3(256), 0, stream>>>(emb, idx, zq);
}
